// MsPoELlamaAttention_20504173871153
// MI455X (gfx1250) — matrix operand order repaired and checked by reading, not compiled, not run on hardware
//
#include <hip/hip_runtime.h>
#include <hip/hip_bf16.h>
#include <math.h>

#define H_    32
#define D_    128
#define HID_  4096
#define KT    32
#define LDA   48          // padded LDS row stride in halves (96B -> 16B-aligned frags)
#define LN10000 9.210340371976184f

typedef __attribute__((ext_vector_type(16))) _Float16 v16h;
typedef __attribute__((ext_vector_type(8)))  _Float16 v8h;
typedef __attribute__((ext_vector_type(8)))  float    v8f;

static __device__ __forceinline__ v16h cat16(v8h lo, v8h hi) {
  return __builtin_shufflevector(lo, hi, 0,1,2,3,4,5,6,7,8,9,10,11,12,13,14,15);
}
static __device__ __forceinline__ v8f wmma_f16(v16h a, v16h b, v8f c) {
  // D = A(16x32 f16) x B(32x16 f16) + C(16x16 f32)
  return __builtin_amdgcn_wmma_f32_16x16x32_f16(false, a, false, b, (short)0, c, false, false);
}

// ---------------------------------------------------------------------------
// Kernel 1: fused QKV projection + per-head MsPoE RoPE.
// grid = (S/128, HID/128 = heads, 3{q,k,v}), block = 256 (8 waves).
// Double-buffered LDS + register prefetch of the next K-tile; B fragments
// ping-pong through TWO register octets (overlapping live ranges force two
// physical blocks) so each ds_load pair hides behind a WMMA (dscnt<=2).
// Wave w computes rows [w*16,w*16+16) x the full 128-wide head, so RoPE
// pairs (d,d+64) are tiles t and t+4 in the SAME lanes (register-only RoPE).
// Outputs: q,k f16 [H][S][D] (RoPE applied), v f16 transposed [H][D][S].
// ---------------------------------------------------------------------------
__global__ __launch_bounds__(256) void qkv_rope_kernel(
    const float* __restrict__ hs, const int* __restrict__ pos,
    const float* __restrict__ wq, const float* __restrict__ wk,
    const float* __restrict__ wv,
    _Float16* __restrict__ qb, _Float16* __restrict__ kb,
    _Float16* __restrict__ vt, int S)
{
  __shared__ _Float16 Asm[2][128 * LDA];
  __shared__ _Float16 Bsm[2][128 * LDA];

  const int tid  = threadIdx.x;
  const int wave = tid >> 5, lane = tid & 31;
  const int lh   = lane & 15, sel = lane >> 4;
  const int s0   = blockIdx.x * 128;
  const int head = blockIdx.y;
  const int n0   = head * 128;
  const int which = blockIdx.z;
  const float* w = (which == 0) ? wq : ((which == 1) ? wk : wv);
  const int m0 = wave * 16;

  const v8f zero = {0.f,0.f,0.f,0.f,0.f,0.f,0.f,0.f};
  v8f acc[8];
#pragma unroll
  for (int t = 0; t < 8; ++t) acc[t] = zero;

  float4 ra[4], rb[4];
  // prologue: fetch tile k0=0 into registers
#pragma unroll
  for (int i = 0; i < 4; ++i) {
    const int idx = tid + i * 256;
    const int row = idx >> 3;
    const int c4  = (idx & 7) << 2;
    ra[i] = *(const float4*)(hs + (size_t)(s0 + row) * HID_ + c4);
    rb[i] = *(const float4*)(w  + (size_t)(n0 + row) * HID_ + c4);
  }

  int buf = 0;
  for (int k0 = 0; k0 < HID_; k0 += KT, buf ^= 1) {
    // stage current tile regs -> LDS (f32 -> f16)
#pragma unroll
    for (int i = 0; i < 4; ++i) {
      const int idx = tid + i * 256;
      const int row = idx >> 3;
      const int c4  = (idx & 7) << 2;
      _Float16* da = &Asm[buf][row * LDA + c4];
      da[0] = (_Float16)ra[i].x; da[1] = (_Float16)ra[i].y;
      da[2] = (_Float16)ra[i].z; da[3] = (_Float16)ra[i].w;
      _Float16* db = &Bsm[buf][row * LDA + c4];
      db[0] = (_Float16)rb[i].x; db[1] = (_Float16)rb[i].y;
      db[2] = (_Float16)rb[i].z; db[3] = (_Float16)rb[i].w;
    }
    __syncthreads();
    // issue next tile's global loads; they overlap the WMMA burst below
    if (k0 + KT < HID_) {
#pragma unroll
      for (int i = 0; i < 4; ++i) {
        const int idx = tid + i * 256;
        const int row = idx >> 3;
        const int c4  = (idx & 7) << 2;
        ra[i] = *(const float4*)(hs + (size_t)(s0 + row) * HID_ + k0 + KT + c4);
        rb[i] = *(const float4*)(w  + (size_t)(n0 + row) * HID_ + k0 + KT + c4);
      }
    }
    // A fragment: lanes 0-15 K{0..7,16..23}, lanes 16-31 K{8..15,24..31}
    const _Float16* ap = &Asm[buf][(m0 + lh) * LDA];
    const v16h a = cat16(*(const v8h*)(ap + sel * 8),
                         *(const v8h*)(ap + 16 + sel * 8));
    // ping-pong B pipeline over two octets
    auto loadB = [&](int t) -> v16h {
      const _Float16* bp = &Bsm[buf][(t * 16 + lh) * LDA + sel * 8];
      return cat16(*(const v8h*)bp, *(const v8h*)(bp + 16));
    };
    v16h b0 = loadB(0);
    v16h b1 = loadB(1);
#pragma unroll
    for (int t = 0; t < 8; t += 2) {
      acc[t] = wmma_f16(a, b0, acc[t]);
      if (t + 2 < 8) b0 = loadB(t + 2);
      acc[t + 1] = wmma_f16(a, b1, acc[t + 1]);
      if (t + 3 < 8) b1 = loadB(t + 3);
    }
  }

  if (which < 2) {
    // RoPE epilogue: head uses reversed index hh = H-1-head.
    _Float16* out = (which == 0) ? qb : kb;
    const int hh = H_ - 1 - head;
    const float ratio = 1.0f + 2.0f * (float)hh / (float)H_;   // MIN_R..MAX_R
#pragma unroll
    for (int t = 0; t < 4; ++t) {
      const int j = t * 16 + lh;                               // d % 64
      const float invf = __expf(-(float)(2 * j) * (LN10000 / (float)D_));
#pragma unroll
      for (int r = 0; r < 8; ++r) {
        const int srow = s0 + m0 + r + sel * 8;
        const float p = (float)pos[srow];
        float sn, cs;
        __sincosf((p / ratio) * invf, &sn, &cs);
        const float lo = acc[t][r], hi = acc[t + 4][r];
        const size_t base = ((size_t)head * S + srow) * (size_t)D_;
        out[base + j]      = (_Float16)(lo * cs - hi * sn);
        out[base + 64 + j] = (_Float16)(hi * cs + lo * sn);
      }
    }
  } else {
    // v: store transposed [H][D][S] so PV B-fragments are contiguous over keys
#pragma unroll
    for (int t = 0; t < 8; ++t)
#pragma unroll
      for (int r = 0; r < 8; ++r) {
        const int srow = s0 + m0 + r + sel * 8;
        const int d = t * 16 + lh;
        vt[((size_t)head * D_ + d) * S + srow] = (_Float16)acc[t][r];
      }
  }
}

// ---------------------------------------------------------------------------
// Kernel 2: causal flash attention per head.
// grid = (S/64, H), block = 128 (4 waves, each owns 16 query rows).
// ---------------------------------------------------------------------------
__global__ __launch_bounds__(128) void attn_kernel(
    const _Float16* __restrict__ qb, const _Float16* __restrict__ kb,
    const _Float16* __restrict__ vt, _Float16* __restrict__ ao, int S)
{
  __shared__ _Float16 Plds[4 * 16 * LDA];   // per-wave 16x32 P staging (C->A layout)

  const int tid  = threadIdx.x;
  const int wave = tid >> 5, lane = tid & 31;
  const int lh   = lane & 15, sel = lane >> 4;
  const int head  = blockIdx.y;
  const int qrow0 = blockIdx.x * 64 + wave * 16;

  const _Float16* qh = qb + (size_t)head * S * D_;
  const _Float16* kh = kb + (size_t)head * S * D_;
  const _Float16* vh = vt + (size_t)head * D_ * S;

  // Q fragments for all 4 K-chunks of D=128
  v16h qa[4];
  {
    const _Float16* qp = qh + (size_t)(qrow0 + lh) * D_;
#pragma unroll
    for (int c = 0; c < 4; ++c)
      qa[c] = cat16(*(const v8h*)(qp + c * 32 + sel * 8),
                    *(const v8h*)(qp + c * 32 + 16 + sel * 8));
  }

  const v8f zero = {0.f,0.f,0.f,0.f,0.f,0.f,0.f,0.f};
  float mstat[8], lstat[8];
  v8f o[8];
#pragma unroll
  for (int r = 0; r < 8; ++r) { mstat[r] = -1e30f; lstat[r] = 0.f; }
#pragma unroll
  for (int dt = 0; dt < 8; ++dt) o[dt] = zero;

  const float scale = 0.08838834764831845f;   // 1/sqrt(128)
  const int nkb = (qrow0 + 47) >> 5;          // key blocks of 32 covering keys<=row

  for (int kbi = 0; kbi < nkb; ++kbi) {
    const int kbase = kbi * 32;
    if (kbi + 1 < nkb) {   // prefetch next key block (K rows, V rows)
      __builtin_prefetch(kh + (size_t)(kbase + 32 + lane) * D_, 0, 1);
      __builtin_prefetch(vh + (size_t)(lane * 4) * S + kbase + 32, 0, 1);
    }
    // S = Q K^T: ping-pong pipeline over 8 (nt,c) fragments
    v8f sacc[2] = { zero, zero };
    {
      auto loadK = [&](int i) -> v16h {
        const int nt = i >> 2, c = i & 3;
        const _Float16* kp =
            kh + (size_t)(kbase + nt * 16 + lh) * D_ + c * 32 + sel * 8;
        return cat16(*(const v8h*)kp, *(const v8h*)(kp + 16));
      };
      v16h f0 = loadK(0);
      v16h f1 = loadK(1);
#pragma unroll
      for (int i = 0; i < 8; i += 2) {
        sacc[i >> 2] = wmma_f16(qa[i & 3], f0, sacc[i >> 2]);
        if (i + 2 < 8) f0 = loadK(i + 2);
        sacc[(i + 1) >> 2] = wmma_f16(qa[(i + 1) & 3], f1, sacc[(i + 1) >> 2]);
        if (i + 3 < 8) f1 = loadK(i + 3);
      }
    }

    const bool diag = (kbi == nkb - 1);
    float mloc[8];
#pragma unroll
    for (int r = 0; r < 8; ++r) {
      float s0v = sacc[0][r] * scale;
      float s1v = sacc[1][r] * scale;
      if (diag) {
        const int row = qrow0 + r + sel * 8;
        if (kbase + lh      > row) s0v = -1e30f;
        if (kbase + 16 + lh > row) s1v = -1e30f;
      }
      sacc[0][r] = s0v; sacc[1][r] = s1v;
      float mv = fmaxf(s0v, s1v);
#pragma unroll
      for (int m = 1; m <= 8; m <<= 1) mv = fmaxf(mv, __shfl_xor(mv, m, 32));
      mloc[r] = mv;
    }

#pragma unroll
    for (int r = 0; r < 8; ++r) {
      const float mnew  = fmaxf(mstat[r], mloc[r]);
      const float alpha = __expf(mstat[r] - mnew);
      const float p0 = __expf(sacc[0][r] - mnew);
      const float p1 = __expf(sacc[1][r] - mnew);
      float rs = p0 + p1;
#pragma unroll
      for (int m = 1; m <= 8; m <<= 1) rs += __shfl_xor(rs, m, 32);
      lstat[r] = lstat[r] * alpha + rs;
      mstat[r] = mnew;
#pragma unroll
      for (int dt = 0; dt < 8; ++dt) o[dt][r] *= alpha;
      // stage P (C layout -> LDS row-major)
      _Float16* pl = &Plds[wave * 16 * LDA + (r + sel * 8) * LDA];
      pl[lh]      = (_Float16)p0;
      pl[16 + lh] = (_Float16)p1;
    }
    asm volatile("s_wait_dscnt 0" ::: "memory");   // per-wave DS RAW fence

    // reload P as A-fragment (16x32)
    const _Float16* pp = &Plds[wave * 16 * LDA + lh * LDA];
    const v16h pa = cat16(*(const v8h*)(pp + sel * 8),
                          *(const v8h*)(pp + 16 + sel * 8));

    // O += P x V (V transposed [D][S]); ping-pong pipeline over 8 d-tiles
    {
      auto loadV = [&](int dt) -> v16h {
        const _Float16* vp = vh + (size_t)(dt * 16 + lh) * S + kbase + sel * 8;
        return cat16(*(const v8h*)vp, *(const v8h*)(vp + 16));
      };
      v16h f0 = loadV(0);
      v16h f1 = loadV(1);
#pragma unroll
      for (int dt = 0; dt < 8; dt += 2) {
        o[dt] = wmma_f16(pa, f0, o[dt]);
        if (dt + 2 < 8) f0 = loadV(dt + 2);
        o[dt + 1] = wmma_f16(pa, f1, o[dt + 1]);
        if (dt + 3 < 8) f1 = loadV(dt + 3);
      }
    }
  }

  // normalize and store to [S][h*D+d] f16 (input layout for out-projection)
#pragma unroll
  for (int dt = 0; dt < 8; ++dt)
#pragma unroll
    for (int r = 0; r < 8; ++r) {
      const int row = qrow0 + r + sel * 8;
      const int col = head * D_ + dt * 16 + lh;
      ao[(size_t)row * HID_ + col] = (_Float16)(o[dt][r] / lstat[r]);
    }
}

// ---------------------------------------------------------------------------
// Kernel 3: output projection  out = attn @ wo.T  (f16 x f16 -> f32)
// grid = (S/128, HID/128), block = 256 (8 waves). Same pipelined skeleton.
// ---------------------------------------------------------------------------
__global__ __launch_bounds__(256) void proj_kernel(
    const _Float16* __restrict__ ao, const float* __restrict__ wo,
    float* __restrict__ out, int S)
{
  __shared__ _Float16 Asm[2][128 * LDA];
  __shared__ _Float16 Bsm[2][128 * LDA];

  const int tid  = threadIdx.x;
  const int wave = tid >> 5, lane = tid & 31;
  const int lh   = lane & 15, sel = lane >> 4;
  const int s0 = blockIdx.x * 128;
  const int n0 = blockIdx.y * 128;
  const int m0 = wave * 16;

  const v8f zero = {0.f,0.f,0.f,0.f,0.f,0.f,0.f,0.f};
  v8f acc[8];
#pragma unroll
  for (int t = 0; t < 8; ++t) acc[t] = zero;

  v8h raf[2];
  float4 rbf[4];
#pragma unroll
  for (int i = 0; i < 2; ++i) {
    const int idx = tid + i * 256;      // 0..511
    const int row = idx >> 2;
    const int cs8 = (idx & 3) * 8;
    raf[i] = *(const v8h*)(ao + (size_t)(s0 + row) * HID_ + cs8);
  }
#pragma unroll
  for (int i = 0; i < 4; ++i) {
    const int idx = tid + i * 256;
    const int row = idx >> 3;
    const int c4  = (idx & 7) << 2;
    rbf[i] = *(const float4*)(wo + (size_t)(n0 + row) * HID_ + c4);
  }

  int buf = 0;
  for (int k0 = 0; k0 < HID_; k0 += KT, buf ^= 1) {
#pragma unroll
    for (int i = 0; i < 2; ++i) {
      const int idx = tid + i * 256;
      const int row = idx >> 2;
      const int cs8 = (idx & 3) * 8;
      *(v8h*)&Asm[buf][row * LDA + cs8] = raf[i];
    }
#pragma unroll
    for (int i = 0; i < 4; ++i) {
      const int idx = tid + i * 256;
      const int row = idx >> 3;
      const int c4  = (idx & 7) << 2;
      _Float16* db = &Bsm[buf][row * LDA + c4];
      db[0] = (_Float16)rbf[i].x; db[1] = (_Float16)rbf[i].y;
      db[2] = (_Float16)rbf[i].z; db[3] = (_Float16)rbf[i].w;
    }
    __syncthreads();
    if (k0 + KT < HID_) {
#pragma unroll
      for (int i = 0; i < 2; ++i) {
        const int idx = tid + i * 256;
        const int row = idx >> 2;
        const int cs8 = (idx & 3) * 8;
        raf[i] = *(const v8h*)(ao + (size_t)(s0 + row) * HID_ + k0 + KT + cs8);
      }
#pragma unroll
      for (int i = 0; i < 4; ++i) {
        const int idx = tid + i * 256;
        const int row = idx >> 3;
        const int c4  = (idx & 7) << 2;
        rbf[i] = *(const float4*)(wo + (size_t)(n0 + row) * HID_ + k0 + KT + c4);
      }
    }
    const _Float16* ap = &Asm[buf][(m0 + lh) * LDA];
    const v16h a = cat16(*(const v8h*)(ap + sel * 8),
                         *(const v8h*)(ap + 16 + sel * 8));
    auto loadB = [&](int t) -> v16h {
      const _Float16* bp = &Bsm[buf][(t * 16 + lh) * LDA + sel * 8];
      return cat16(*(const v8h*)bp, *(const v8h*)(bp + 16));
    };
    v16h b0 = loadB(0);
    v16h b1 = loadB(1);
#pragma unroll
    for (int t = 0; t < 8; t += 2) {
      acc[t] = wmma_f16(a, b0, acc[t]);
      if (t + 2 < 8) b0 = loadB(t + 2);
      acc[t + 1] = wmma_f16(a, b1, acc[t + 1]);
      if (t + 3 < 8) b1 = loadB(t + 3);
    }
  }

#pragma unroll
  for (int t = 0; t < 8; ++t)
#pragma unroll
    for (int r = 0; r < 8; ++r) {
      const int row = s0 + m0 + r + sel * 8;
      out[(size_t)row * HID_ + n0 + t * 16 + lh] = acc[t][r];
    }
}

// ---------------------------------------------------------------------------
extern "C" void kernel_launch(void* const* d_in, const int* in_sizes, int n_in,
                              void* d_out, int out_size, void* d_ws, size_t ws_size,
                              hipStream_t stream) {
  const float* hs  = (const float*)d_in[0];
  const int*   pos = (const int*)  d_in[1];
  const float* wq  = (const float*)d_in[2];
  const float* wk  = (const float*)d_in[3];
  const float* wv  = (const float*)d_in[4];
  const float* wo  = (const float*)d_in[5];
  const int S = in_sizes[0] / HID_;          // 2048 (B=1)

  const size_t per = (size_t)H_ * S * D_;    // 8M halves each
  _Float16* qb = (_Float16*)d_ws;
  _Float16* kb = qb + per;
  _Float16* vt = kb + per;
  _Float16* ao = vt + per;

  dim3 g1(S / 128, HID_ / 128, 3);
  qkv_rope_kernel<<<g1, 256, 0, stream>>>(hs, pos, wq, wk, wv, qb, kb, vt, S);

  dim3 g2(S / 64, H_);
  attn_kernel<<<g2, 128, 0, stream>>>(qb, kb, vt, ao, S);

  dim3 g3(S / 128, HID_ / 128);
  proj_kernel<<<g3, 256, 0, stream>>>(ao, wo, (float*)d_out, S);
}
